// ShortLongMixLayer_57904749085215
// MI455X (gfx1250) — matrix operand order repaired and checked by reading, not compiled, not run on hardware
//
#include <hip/hip_runtime.h>
#include <hip/hip_bf16.h>

typedef _Float16 f16;
typedef __attribute__((ext_vector_type(16))) _Float16 v16h;
typedef __attribute__((ext_vector_type(4)))  _Float16 v4h;
typedef __attribute__((ext_vector_type(8)))  float    v8f;
typedef __attribute__((ext_vector_type(4)))  float    v4f;

#define NN 50000
#define MM 16384
#define EE 400000
#define TT 1000000
#define HH 128

__device__ __forceinline__ float siluf(float x) {
    return x * __builtin_amdgcn_rcpf(1.0f + __expf(-x));
}

__device__ __forceinline__ float wave_sum(float v) {
#pragma unroll
    for (int o = 16; o > 0; o >>= 1) v += __shfl_xor(v, o, 32);
    return v;
}

__device__ __forceinline__ v8f wmma_f16(v16h a, v16h b, v8f c) {
    return __builtin_amdgcn_wmma_f32_16x16x32_f16(false, a, false, b, (short)0, c, false, false);
}

// ---------------- weight conversion ----------------
// W [K][Ncol] f32 row-major  ->  out [Ncol][K] f16 (column-gather friendly)
__global__ void wt_f16_kernel(const float* __restrict__ W, f16* __restrict__ out,
                              int K, int Ncol) {
    int i = blockIdx.x * blockDim.x + threadIdx.x;
    int total = K * Ncol;
    if (i >= total) return;
    int k = i / Ncol, n = i - k * Ncol;
    out[(size_t)n * K + k] = (f16)W[i];
}

// conv_k [O=128][I=128][27] -> out [27][O][I] f16
__global__ void convw_f16_kernel(const float* __restrict__ Wk, f16* __restrict__ out) {
    int i = blockIdx.x * blockDim.x + threadIdx.x;
    if (i >= 128 * 128 * 27) return;
    int tap = i % 27;
    int rest = i / 27;
    int ii = rest % 128;
    int o  = rest / 128;
    out[((size_t)tap * 128 + o) * 128 + ii] = (f16)Wk[i];
}

// ---------------- LayerNorm (+ optional post-adds) ----------------
__global__ void ln_add_kernel(const float* __restrict__ x, const float* __restrict__ g,
                              const float* __restrict__ b, const float* __restrict__ add1,
                              const float* __restrict__ add2, float* __restrict__ y,
                              int rows) {
    int wave = (int)((blockIdx.x * blockDim.x + threadIdx.x) >> 5);
    int lane = threadIdx.x & 31;
    if (wave >= rows) return;
    const float* xr = x + (size_t)wave * HH;
    float v[4];
    float s = 0.f;
#pragma unroll
    for (int i = 0; i < 4; ++i) { v[i] = xr[lane + 32 * i]; s += v[i]; }
    float mu = wave_sum(s) * (1.0f / HH);
    float var = 0.f;
#pragma unroll
    for (int i = 0; i < 4; ++i) { float d = v[i] - mu; var += d * d; }
    var = wave_sum(var) * (1.0f / HH);
    float rs = rsqrtf(var + 1e-5f);
    float* yr = y + (size_t)wave * HH;
#pragma unroll
    for (int i = 0; i < 4; ++i) {
        int col = lane + 32 * i;
        float o = (v[i] - mu) * rs * g[col] + b[col];
        if (add1) o += add1[(size_t)wave * HH + col];
        if (add2) o += add2[(size_t)wave * HH + col];
        yr[col] = o;
    }
}

// ---------------- triplet interaction: basis * gather, scatter-add ----------------
__global__ void triplet_kernel(const float* __restrict__ rbf3, const float* __restrict__ cbf3,
                               const int* __restrict__ id3_ba, const int* __restrict__ id3_ca,
                               const float* __restrict__ me, const float* __restrict__ W_rbf,
                               const float* __restrict__ W_cbf, float* __restrict__ agg) {
    __shared__ float sWr[16 * HH];
    __shared__ float sWc[16 * HH];
    for (int i = threadIdx.x; i < 16 * HH; i += blockDim.x) {
        sWr[i] = W_rbf[i];
        sWc[i] = W_cbf[i];
    }
    __syncthreads();
    int lane = threadIdx.x & 31;
    int gw = (int)((blockIdx.x * blockDim.x + threadIdx.x) >> 5);
    int nw = (int)((gridDim.x * blockDim.x) >> 5);
    for (int t = gw; t < TT; t += nw) {
        const float* rp = rbf3 + (size_t)t * 16;
        const float* cp = cbf3 + (size_t)t * 16;
        float r[16], c[16];
#pragma unroll
        for (int k = 0; k < 16; ++k) { r[k] = rp[k]; c[k] = cp[k]; }
        int ba = id3_ba[t], ca = id3_ca[t];
        const float* merow = me + (size_t)ba * HH;
        float* aggrow = agg + (size_t)ca * HH;
#pragma unroll
        for (int j = 0; j < 4; ++j) {
            int col = lane + 32 * j;
            float br = 0.f, bc = 0.f;
#pragma unroll
            for (int k = 0; k < 16; ++k) {
                br += r[k] * sWr[k * HH + col];
                bc += c[k] * sWc[k * HH + col];
            }
            atomicAdd(&aggrow[col], merow[col] * br * bc);
        }
    }
}

// ---------------- WMMA GEMM: out = res + silu(A[rows,128] @ W[128,128]) ----------------
// WT f16 [n][k]. Block 256 thr = 8 waves; 32 rows x 128 cols; each wave: 16 rows x 32 cols
// (two 16x16 tiles sharing one A fragment -> two independent WMMA chains).
__global__ void __launch_bounds__(256)
gemm128_silu_kernel(const float* __restrict__ A, const f16* __restrict__ WT,
                    const float* __restrict__ res, float* __restrict__ out, int rows) {
    __shared__ f16 sA[32 * HH];
    int row0 = blockIdx.x * 32;
    int tid = threadIdx.x;
    // staged f32->f16 conversion, float4 loads
    for (int idx = tid; idx < 32 * HH / 4; idx += 256) {
        int r = idx >> 5, c4 = idx & 31;
        int gr = row0 + r;
        v4f x = {};
        if (gr < rows) x = *(const v4f*)(A + (size_t)gr * HH + c4 * 4);
        v4h h;
#pragma unroll
        for (int i = 0; i < 4; ++i) h[i] = (f16)x[i];
        *((v4h*)sA + idx) = h;
    }
    __syncthreads();
    int wave = tid >> 5, lane = tid & 31;
    int rg = wave >> 2;           // row group (0/1)
    int n0 = (wave & 3) * 32;     // col group base
    int arow = lane & 15, kh = lane >> 4;
    const f16* sArow = sA + (rg * 16 + arow) * HH;
    v8f c0 = {}, c1 = {};
#pragma unroll
    for (int kc = 0; kc < 4; ++kc) {
        v16h a;
#pragma unroll
        for (int i = 0; i < 8; ++i) {
            a[i]     = sArow[kc * 32 + kh * 16 + i];
            a[i + 8] = sArow[kc * 32 + kh * 16 + 8 + i];
        }
        v16h b0 = *(const v16h*)(WT + (size_t)(n0 + arow) * HH + kc * 32 + kh * 16);
        v16h b1 = *(const v16h*)(WT + (size_t)(n0 + 16 + arow) * HH + kc * 32 + kh * 16);
        c0 = wmma_f16(a, b0, c0);
        c1 = wmma_f16(a, b1, c1);
    }
#pragma unroll
    for (int v = 0; v < 8; ++v) {
        int mrow = row0 + rg * 16 + kh * 8 + v;
        if (mrow < rows) {
            size_t o0 = (size_t)mrow * HH + n0 + arow;
            out[o0]      = res[o0]      + siluf(c0[v]);
            out[o0 + 16] = res[o0 + 16] + siluf(c1[v]);
        }
    }
}

// ---------------- swap-symmetrize + hmsg scatter ----------------
__global__ void swap_hmsg_kernel(const float* __restrict__ me1, const int* __restrict__ id_swap,
                                 const float* __restrict__ rbf_h, const float* __restrict__ Wh,
                                 const int* __restrict__ idx_t, float* __restrict__ me2,
                                 float* __restrict__ aagg) {
    __shared__ float sW[16 * HH];
    for (int i = threadIdx.x; i < 16 * HH; i += blockDim.x) sW[i] = Wh[i];
    __syncthreads();
    int lane = threadIdx.x & 31;
    int gw = (int)((blockIdx.x * blockDim.x + threadIdx.x) >> 5);
    int nw = (int)((gridDim.x * blockDim.x) >> 5);
    for (int e = gw; e < EE; e += nw) {
        int sw = id_swap[e], t = idx_t[e];
        const float* r1 = me1 + (size_t)e * HH;
        const float* r2 = me1 + (size_t)sw * HH;
        const float* rp = rbf_h + (size_t)e * 16;
        float rb[16];
#pragma unroll
        for (int k = 0; k < 16; ++k) rb[k] = rp[k];
        float* o = me2 + (size_t)e * HH;
        float* ag = aagg + (size_t)t * HH;
#pragma unroll
        for (int j = 0; j < 4; ++j) {
            int col = lane + 32 * j;
            float v = r1[col] + r2[col];
            o[col] = v;
            float p = 0.f;
#pragma unroll
            for (int k = 0; k < 16; ++k) p += rb[k] * sW[k * HH + col];
            atomicAdd(&ag[col], v * p);
        }
    }
}

// ---------------- generic edge message scatter ----------------
__global__ void edge_scatter_kernel(const float* __restrict__ src, const int* __restrict__ eidx,
                                    const float* __restrict__ attr, const float* __restrict__ W,
                                    const float* __restrict__ ew, float* __restrict__ dst) {
    __shared__ float sW[16 * HH];
    for (int i = threadIdx.x; i < 16 * HH; i += blockDim.x) sW[i] = W[i];
    __syncthreads();
    int lane = threadIdx.x & 31;
    int gw = (int)((blockIdx.x * blockDim.x + threadIdx.x) >> 5);
    int nw = (int)((gridDim.x * blockDim.x) >> 5);
    for (int e = gw; e < EE; e += nw) {
        int s = eidx[e], t = eidx[EE + e];
        float wgt = ew[e];
        const float* sr = src + (size_t)s * HH;
        const float* ap = attr + (size_t)e * 16;
        float at[16];
#pragma unroll
        for (int k = 0; k < 16; ++k) at[k] = ap[k];
        float* d = dst + (size_t)t * HH;
#pragma unroll
        for (int j = 0; j < 4; ++j) {
            int col = lane + 32 * j;
            float p = 0.f;
#pragma unroll
            for (int k = 0; k < 16; ++k) p += at[k] * sW[k * HH + col];
            atomicAdd(&d[col], sr[col] * siluf(p) * wgt);
        }
    }
}

// ---------------- 3D conv (3x3x3 SAME, 128->128 ch): 27 WMMA-accumulated taps ----------------
__global__ void __launch_bounds__(256)
conv3d_kernel(const float* __restrict__ xin, const f16* __restrict__ Wc,
              float* __restrict__ mx) {
    __shared__ f16 sA[16 * HH];
    int row0 = blockIdx.x * 16;  // 16 voxels per block (same b,d; h0,h0+1, all w)
    int tid = threadIdx.x;
    int wave = tid >> 5, lane = tid & 31;
    int arow = lane & 15, kh2 = lane >> 4;
    int n0 = wave * 16;
    v8f c = {};
    for (int tap = 0; tap < 27; ++tap) {
        int kd = tap / 9, khh = (tap / 3) % 3, kw = tap % 3;
        __syncthreads();
        for (int idx = tid; idx < 16 * HH / 4; idx += 256) {
            int r = idx >> 5, c4 = idx & 31;
            int mvx = row0 + r;
            int bb = mvx >> 9, rem = mvx & 511;
            int d = rem >> 6, h = (rem >> 3) & 7, w = rem & 7;
            int dd = d + kd - 1, hh = h + khh - 1, ww = w + kw - 1;
            v4f x = {};
            if ((unsigned)dd < 8u && (unsigned)hh < 8u && (unsigned)ww < 8u) {
                int mn = (bb << 9) + (dd << 6) + (hh << 3) + ww;
                x = *(const v4f*)(xin + (size_t)mn * HH + c4 * 4);
            }
            v4h hvec;
#pragma unroll
            for (int i = 0; i < 4; ++i) hvec[i] = (f16)x[i];
            *((v4h*)sA + idx) = hvec;
        }
        __syncthreads();
        const f16* Wt = Wc + (size_t)tap * 128 * 128;
#pragma unroll
        for (int kc = 0; kc < 4; ++kc) {
            v16h a;
#pragma unroll
            for (int i = 0; i < 8; ++i) {
                a[i]     = sA[arow * HH + kc * 32 + kh2 * 16 + i];
                a[i + 8] = sA[arow * HH + kc * 32 + kh2 * 16 + 8 + i];
            }
            v16h b = *(const v16h*)(Wt + (size_t)(n0 + arow) * HH + kc * 32 + kh2 * 16);
            c = wmma_f16(a, b, c);
        }
    }
    int n = n0 + arow;
#pragma unroll
    for (int v = 0; v < 8; ++v) {
        int mrow = kh2 * 8 + v;
        mx[(size_t)(row0 + mrow) * HH + n] = siluf(c[v]);
    }
}

// ---------------- em GEMM: silu(concat(m2a[idx_s], m2a[idx_t]) @ W_comb + b) ----------------
// WT: [128 n][256 k] f16. 32 rows/block; each wave two 16x16 tiles.
__global__ void __launch_bounds__(256)
gemm_comb_kernel(const float* __restrict__ m2a, const int* __restrict__ idx_s,
                 const int* __restrict__ idx_t, const f16* __restrict__ WT,
                 const float* __restrict__ bias, float* __restrict__ out, int rows) {
    __shared__ f16 sA[32 * 256];
    int row0 = blockIdx.x * 32;
    int tid = threadIdx.x;
    for (int idx = tid; idx < 32 * 256 / 4; idx += 256) {
        int r = idx >> 6, c4 = idx & 63;  // 64 float4 per 256-wide row
        int gr = row0 + r;
        v4f x = {};
        if (gr < rows) {
            int src = (c4 < 32) ? idx_s[gr] : idx_t[gr];
            x = *(const v4f*)(m2a + (size_t)src * HH + (c4 & 31) * 4);
        }
        v4h h;
#pragma unroll
        for (int i = 0; i < 4; ++i) h[i] = (f16)x[i];
        *((v4h*)sA + idx) = h;
    }
    __syncthreads();
    int wave = tid >> 5, lane = tid & 31;
    int rg = wave >> 2;
    int n0 = (wave & 3) * 32;
    int arow = lane & 15, kh = lane >> 4;
    const f16* sArow = sA + (rg * 16 + arow) * 256;
    v8f c0 = {}, c1 = {};
#pragma unroll
    for (int kc = 0; kc < 8; ++kc) {
        v16h a;
#pragma unroll
        for (int i = 0; i < 8; ++i) {
            a[i]     = sArow[kc * 32 + kh * 16 + i];
            a[i + 8] = sArow[kc * 32 + kh * 16 + 8 + i];
        }
        v16h b0 = *(const v16h*)(WT + (size_t)(n0 + arow) * 256 + kc * 32 + kh * 16);
        v16h b1 = *(const v16h*)(WT + (size_t)(n0 + 16 + arow) * 256 + kc * 32 + kh * 16);
        c0 = wmma_f16(a, b0, c0);
        c1 = wmma_f16(a, b1, c1);
    }
    int n = n0 + arow;
    float bn0 = bias[n], bn1 = bias[n + 16];
#pragma unroll
    for (int v = 0; v < 8; ++v) {
        int mrow = row0 + rg * 16 + kh * 8 + v;
        if (mrow < rows) {
            size_t o0 = (size_t)mrow * HH + n;
            out[o0]      = siluf(c0[v] + bn0);
            out[o0 + 16] = siluf(c1[v] + bn1);
        }
    }
}

extern "C" void kernel_launch(void* const* d_in, const int* in_sizes, int n_in,
                              void* d_out, int out_size, void* d_ws, size_t ws_size,
                              hipStream_t stream) {
    const float* a_x    = (const float*)d_in[0];
    const float* m_x    = (const float*)d_in[1];
    const float* m_in   = (const float*)d_in[2];
    const float* rbf3   = (const float*)d_in[3];
    const float* cbf3   = (const float*)d_in[4];
    const int*   id_swap= (const int*)d_in[6];
    const int*   id3_ba = (const int*)d_in[7];
    const int*   id3_ca = (const int*)d_in[8];
    const float* rbf_h  = (const float*)d_in[9];
    const int*   idx_s  = (const int*)d_in[10];
    const int*   idx_t  = (const int*)d_in[11];
    const int*   a2m_ei = (const int*)d_in[12];
    const int*   m2a_ei = (const int*)d_in[13];
    const float* a2m_w  = (const float*)d_in[14];
    const float* m2a_w  = (const float*)d_in[15];
    const float* a2m_at = (const float*)d_in[16];
    const float* m2a_at = (const float*)d_in[17];
    const float* W_rbf  = (const float*)d_in[18];
    const float* W_cbf  = (const float*)d_in[19];
    const float* W_m    = (const float*)d_in[20];
    const float* W_h    = (const float*)d_in[21];
    const float* W_a    = (const float*)d_in[22];
    const float* W_a2m  = (const float*)d_in[23];
    const float* W_m2a  = (const float*)d_in[24];
    const float* conv_k = (const float*)d_in[25];
    const float* W_comb = (const float*)d_in[26];
    const float* b_comb = (const float*)d_in[27];
    const float* g_short= (const float*)d_in[28];
    const float* b_short= (const float*)d_in[29];
    const float* g_se   = (const float*)d_in[30];
    const float* b_se   = (const float*)d_in[31];
    const float* g_long = (const float*)d_in[32];
    const float* b_long = (const float*)d_in[33];
    const float* g_a2m  = (const float*)d_in[34];
    const float* b_a2m  = (const float*)d_in[35];
    const float* g_m2ae = (const float*)d_in[36];
    const float* b_m2ae = (const float*)d_in[37];

    const size_t NH = (size_t)NN * HH, EHsz = (size_t)EE * HH, MH = (size_t)MM * HH;
    float* ws   = (float*)d_ws;
    float* aN   = ws;              // [N,128] LN(a_x)
    float* meB  = aN + NH;         // [E,128] LN(m) -> me2 (symmetrized)
    float* ebB  = meB + EHsz;      // [E,128] agg -> me1 -> em_raw
    float* nbB  = ebB + EHsz;      // [N,128] aagg -> m2a_buf
    float* mxnB = nbB + NH;        // [M,128] LN(m_x)
    float* mxB  = mxnB + MH;       // [M,128] conv output (silu'd)
    float* a2mB = mxB + MH;        // [M,128] a2m scatter buffer
    f16* WmT    = (f16*)(a2mB + MH);       // [128][128]
    f16* WaT    = WmT + 128 * 128;         // [128][128]
    f16* WcombT = WaT + 128 * 128;         // [128][256]
    f16* WconvT = WcombT + 128 * 256;      // [27][128][128]

    float* out_a  = (float*)d_out;                // [N,128]
    float* out_m  = out_a + NH;                   // [M,128]
    float* out_me = out_m + MH;                   // [E,128]

    // 1) weight prep
    wt_f16_kernel<<<(128 * 128 + 255) / 256, 256, 0, stream>>>(W_m, WmT, 128, 128);
    wt_f16_kernel<<<(128 * 128 + 255) / 256, 256, 0, stream>>>(W_a, WaT, 128, 128);
    wt_f16_kernel<<<(256 * 128 + 255) / 256, 256, 0, stream>>>(W_comb, WcombT, 256, 128);
    convw_f16_kernel<<<(27 * 128 * 128 + 255) / 256, 256, 0, stream>>>(conv_k, WconvT);

    // 2) pre-norms
    ln_add_kernel<<<(NN + 7) / 8, 256, 0, stream>>>(a_x, g_short, b_short, nullptr, nullptr, aN, NN);
    ln_add_kernel<<<(EE + 7) / 8, 256, 0, stream>>>(m_in, g_se, b_se, nullptr, nullptr, meB, EE);
    ln_add_kernel<<<(MM + 7) / 8, 256, 0, stream>>>(m_x, g_long, b_long, nullptr, nullptr, mxnB, MM);

    // 3) triplet interaction -> agg (ebB)
    hipMemsetAsync(ebB, 0, EHsz * sizeof(float), stream);
    triplet_kernel<<<2048, 256, 0, stream>>>(rbf3, cbf3, id3_ba, id3_ca, meB, W_rbf, W_cbf, ebB);

    // 4) me1 = me + silu(agg @ W_m)   (in-place on ebB; block-local rows)
    gemm128_silu_kernel<<<(EE + 31) / 32, 256, 0, stream>>>(ebB, WmT, meB, ebB, EE);

    // 5) symmetrize + hmsg scatter to atoms
    hipMemsetAsync(nbB, 0, NH * sizeof(float), stream);
    swap_hmsg_kernel<<<2048, 256, 0, stream>>>(ebB, id_swap, rbf_h, W_h, idx_t, meB, nbB);

    // 6) a = aN + silu(aagg @ W_a)  -> output 1
    gemm128_silu_kernel<<<(NN + 31) / 32, 256, 0, stream>>>(nbB, WaT, aN, out_a, NN);

    // 7) long branch: conv3d(LN(m_x)) -> mx (silu'd)
    conv3d_kernel<<<MM / 16, 256, 0, stream>>>(mxnB, WconvT, mxB);

    // 8) a2m scatter, LN, combine -> output 2 = mx + LN(a2m) + m_x
    hipMemsetAsync(a2mB, 0, MH * sizeof(float), stream);
    edge_scatter_kernel<<<2048, 256, 0, stream>>>(out_a, a2m_ei, a2m_at, W_a2m, a2m_w, a2mB);
    ln_add_kernel<<<(MM + 7) / 8, 256, 0, stream>>>(a2mB, g_a2m, b_a2m, mxB, m_x, out_m, MM);

    // 9) m2a scatter into nbB (reuse)
    hipMemsetAsync(nbB, 0, NH * sizeof(float), stream);
    edge_scatter_kernel<<<2048, 256, 0, stream>>>(mxB, m2a_ei, m2a_at, W_m2a, m2a_w, nbB);

    // 10) em = silu(concat @ W_comb + b); out3 = me2 + LN(em)
    gemm_comb_kernel<<<(EE + 31) / 32, 256, 0, stream>>>(nbB, idx_s, idx_t, WcombT, b_comb, ebB, EE);
    ln_add_kernel<<<(EE + 7) / 8, 256, 0, stream>>>(ebB, g_m2ae, b_m2ae, meB, nullptr, out_me, EE);

    (void)in_sizes; (void)n_in; (void)out_size; (void)ws_size;
}
